// APSGNNModel_84310208020969
// MI455X (gfx1250) — matrix operand order repaired and checked by reading, not compiled, not run on hardware
//
#include <hip/hip_runtime.h>
#include <math.h>

// ---------------------------------------------------------------------------
// Types
// ---------------------------------------------------------------------------
typedef __attribute__((ext_vector_type(16))) __bf16 bf16x16;
typedef __attribute__((ext_vector_type(8)))  __bf16 bf16x8;
typedef __attribute__((ext_vector_type(8)))  float  v8f;

#define WMMA_BF16(a, b, c) \
  __builtin_amdgcn_wmma_f32_16x16x32_bf16(false, (a), false, (b), (short)0, (c), false, false)

// round-to-nearest-even f32 -> bf16 (finite inputs)
__device__ __forceinline__ __bf16 f2bf(float x) {
  union { float f; unsigned u; } v; v.f = x;
  unsigned r = v.u + 0x7FFFu + ((v.u >> 16) & 1u);
  union { unsigned short s; __bf16 b; } o; o.s = (unsigned short)(r >> 16);
  return o.b;
}

__device__ __forceinline__ float gelu_exact(float x) {
  return 0.5f * x * (1.f + erff(x * 0.70710678118654752f));
}

// ---------------------------------------------------------------------------
// Fragment loaders
// A: 16x32 bf16 row-major in LDS.  Lanes 0-15: row=lane, K = {0..7,16..23};
//    lanes 16-31: row=lane-16, K = {8..15,24..31}  (ISA 7.12.2 16-bit A layout)
// ---------------------------------------------------------------------------
__device__ __forceinline__ bf16x16 load_frag_A(const __bf16* s, int lda, int k0) {
  int lane = threadIdx.x & 31;
  int row  = lane & 15;
  int kb   = (lane >> 4) << 3;                 // 0 or 8
  const __bf16* p = s + row * lda + k0 + kb;
  bf16x16 f;
  ((bf16x8*)&f)[0] = *(const bf16x8*)(p);      // K = kb .. kb+7
  ((bf16x8*)&f)[1] = *(const bf16x8*)(p + 16); // K = kb+16 .. kb+23
  return f;
}

// B: 32x16 tile pre-packed by prep kernel: lane-major, 16 contiguous bf16/lane.
__device__ __forceinline__ bf16x16 load_frag_B(const __bf16* wmat, int ntiles,
                                               int kt, int nt) {
  int lane = threadIdx.x & 31;
  const __bf16* p = wmat + (((size_t)kt * ntiles + nt) << 9) + lane * 16;
  bf16x16 f;
  ((bf16x8*)&f)[0] = *(const bf16x8*)(p);
  ((bf16x8*)&f)[1] = *(const bf16x8*)(p + 8);
  return f;
}

// ---------------------------------------------------------------------------
// Prep: f32 [K][Nc] row-major -> bf16 fragment-tiled layout in workspace.
// Tile (kt,nt) holds, for lane l, elements W[kt*32 + 8*(l>>4) + (i&7) + 16*(i>>3)][nt*16 + (l&15)]
// ---------------------------------------------------------------------------
__global__ void prep_weights(const float* __restrict__ src, __bf16* __restrict__ dst,
                             int K, int Nc) {
  int ntiles = Nc >> 4;
  int total  = K * Nc;
  for (int d = blockIdx.x * blockDim.x + threadIdx.x; d < total;
       d += gridDim.x * blockDim.x) {
    int tile = d >> 9;
    int l    = (d >> 4) & 31;
    int i    = d & 15;
    int kt   = tile / ntiles, nt = tile % ntiles;
    int k = kt * 32 + ((l >> 4) << 3) + (i & 7) + ((i >> 3) << 4);
    int n = (nt << 4) + (l & 15);
    dst[d] = f2bf(src[(size_t)k * Nc + n]);
  }
}

// ---------------------------------------------------------------------------
// Main fused kernel: 16 rows per block, 256 threads (8 waves)
// ---------------------------------------------------------------------------
__global__ __launch_bounds__(256)
void apsg_main(const float* __restrict__ key, const float* __restrict__ aux,
               const float* __restrict__ res, const int* __restrict__ role,
               const float* __restrict__ bk, const float* __restrict__ ba,
               const float* __restrict__ br,
               const float* __restrict__ g_in, const float* __restrict__ b_in,
               const float* __restrict__ bb0, const float* __restrict__ g0,
               const float* __restrict__ be0,
               const float* __restrict__ bb1, const float* __restrict__ g1,
               const float* __restrict__ be1,
               const float* __restrict__ bw, const float* __restrict__ bq,
               const float* __restrict__ bad,
               const __bf16* __restrict__ wsWk, const __bf16* __restrict__ wsWa,
               const __bf16* __restrict__ wsWr, const __bf16* __restrict__ wsWb0,
               const __bf16* __restrict__ wsWb1, const __bf16* __restrict__ wsWw,
               const __bf16* __restrict__ wsWq, const __bf16* __restrict__ wsWad,
               float* __restrict__ out_logits, float* __restrict__ out_h,
               float* __restrict__ out_addr) {
  __shared__ __align__(16) __bf16 sKey[16 * 128];
  __shared__ __align__(16) __bf16 sAux[16 * 256];
  __shared__ __align__(16) __bf16 sRes[16 * 256];
  __shared__ __align__(16) float  sH[16 * 256];
  __shared__ __align__(16) __bf16 sHb[16 * 256];
  __shared__ int sRole[16];

  const int tid  = threadIdx.x;
  const int wv   = tid >> 5;
  const int lane = tid & 31;
  const int r0   = blockIdx.x * 16;

  // ---- stage inputs to LDS (bf16; residual pre-scaled by 0.1) ----
  for (int i = tid; i < 16 * 128; i += 256) sKey[i] = f2bf(key[(size_t)r0 * 128 + i]);
  for (int i = tid; i < 16 * 256; i += 256) sAux[i] = f2bf(aux[(size_t)r0 * 256 + i]);
  for (int i = tid; i < 16 * 256; i += 256) sRes[i] = f2bf(0.1f * res[(size_t)r0 * 256 + i]);
  if (tid < 16) sRole[tid] = role[r0 + tid];
  __syncthreads();

  const int nhalf = (lane >> 4) << 3;   // row base of this lane's C slots (0 or 8)
  const int ncol  = lane & 15;          // column within tile

  // ---- phase 1: h = key@Wk + aux@Wa + 0.1*res@Wr (+ fused biases) ----
  for (int t = 0; t < 2; ++t) {
    int nt = 2 * wv + t;
    v8f acc = {};
#pragma unroll
    for (int kt = 0; kt < 4; ++kt)
      acc = WMMA_BF16(load_frag_A(sKey, 128, kt * 32), load_frag_B(wsWk, 16, kt, nt), acc);
#pragma unroll
    for (int kt = 0; kt < 8; ++kt)
      acc = WMMA_BF16(load_frag_A(sAux, 256, kt * 32), load_frag_B(wsWa, 16, kt, nt), acc);
#pragma unroll
    for (int kt = 0; kt < 8; ++kt)
      acc = WMMA_BF16(load_frag_A(sRes, 256, kt * 32), load_frag_B(wsWr, 16, kt, nt), acc);
    int n = nt * 16 + ncol;
    float bias = bk[n] + ba[n] + 0.1f * br[n];
#pragma unroll
    for (int r = 0; r < 8; ++r) sH[(nhalf + r) * 256 + n] = acc[r] + bias;
  }
  __syncthreads();

  // ---- LayerNorm helper (wave w owns rows 2w, 2w+1) ----
  auto layernorm = [&](const float* g, const float* b) {
#pragma unroll
    for (int rr = 0; rr < 2; ++rr) {
      int row = 2 * wv + rr;
      float vals[8];
      float s = 0.f;
      const float* x = sH + row * 256 + lane * 8;
#pragma unroll
      for (int j = 0; j < 8; ++j) { vals[j] = x[j]; s += vals[j]; }
#pragma unroll
      for (int o = 16; o; o >>= 1) s += __shfl_xor(s, o, 32);
      float mean = s * (1.f / 256.f);
      float s2 = 0.f;
#pragma unroll
      for (int j = 0; j < 8; ++j) { float d = vals[j] - mean; s2 += d * d; }
#pragma unroll
      for (int o = 16; o; o >>= 1) s2 += __shfl_xor(s2, o, 32);
      float rstd = rsqrtf(s2 * (1.f / 256.f) + 1e-5f);
#pragma unroll
      for (int j = 0; j < 8; ++j) {
        int c = lane * 8 + j;
        float y = (vals[j] - mean) * rstd * g[c] + b[c];
        sH[row * 256 + c]  = y;
        sHb[row * 256 + c] = f2bf(y);
      }
    }
  };

  layernorm(g_in, b_in);
  __syncthreads();

  // ---- backbone: 2 x [Linear -> GELU -> LN] ----
  for (int l = 0; l < 2; ++l) {
    const __bf16* wsB = l ? wsWb1 : wsWb0;
    const float* bb = l ? bb1 : bb0;
    for (int t = 0; t < 2; ++t) {
      int nt = 2 * wv + t;
      v8f acc = {};
#pragma unroll
      for (int kt = 0; kt < 8; ++kt)
        acc = WMMA_BF16(load_frag_A(sHb, 256, kt * 32), load_frag_B(wsB, 16, kt, nt), acc);
      int n = nt * 16 + ncol;
      float bias = bb[n];
#pragma unroll
      for (int r = 0; r < 8; ++r)
        sH[(nhalf + r) * 256 + n] = gelu_exact(acc[r] + bias);
    }
    __syncthreads();
    layernorm(l ? g1 : g0, l ? be1 : be0);
    __syncthreads();
  }

  // ---- store h (final, post-LN) ----
  for (int i = tid; i < 16 * 256; i += 256) out_h[(size_t)r0 * 256 + i] = sH[i];

  // ---- logits head: both Ww and Wq per tile, per-row role select ----
  for (int t = 0; t < 8; ++t) {
    int nt = wv * 8 + t;
    v8f aw = {}, aq = {};
#pragma unroll
    for (int kt = 0; kt < 8; ++kt) {
      bf16x16 a = load_frag_A(sHb, 256, kt * 32);
      aw = WMMA_BF16(a, load_frag_B(wsWw, 64, kt, nt), aw);
      aq = WMMA_BF16(a, load_frag_B(wsWq, 64, kt, nt), aq);
    }
    int n = nt * 16 + ncol;
    float biasW = bw[n], biasQ = bq[n];
#pragma unroll
    for (int r = 0; r < 8; ++r) {
      int rr = nhalf + r;
      float v = (sRole[rr] == 0) ? (aw[r] + biasW) : (aq[r] + biasQ);
      out_logits[(size_t)(r0 + rr) * 1024 + n] = v;
    }
  }

  // ---- address head (4 tiles) on waves 4..7 (wave-uniform branch) ----
  if (wv >= 4) {
    int nt = wv - 4;
    v8f ac = {};
#pragma unroll
    for (int kt = 0; kt < 8; ++kt)
      ac = WMMA_BF16(load_frag_A(sHb, 256, kt * 32), load_frag_B(wsWad, 4, kt, nt), ac);
    int n = nt * 16 + ncol;
    float bias = bad[n];
#pragma unroll
    for (int r = 0; r < 8; ++r)
      out_addr[(size_t)(r0 + nhalf + r) * 64 + n] = ac[r] + bias;
  }
}

// ---------------------------------------------------------------------------
// Launch
// ---------------------------------------------------------------------------
extern "C" void kernel_launch(void* const* d_in, const int* in_sizes, int n_in,
                              void* d_out, int out_size, void* d_ws, size_t ws_size,
                              hipStream_t stream) {
  (void)in_sizes; (void)n_in; (void)out_size; (void)ws_size;
  const int N = 131072;

  const float* key  = (const float*)d_in[0];
  const float* aux  = (const float*)d_in[1];
  const float* res  = (const float*)d_in[2];
  const int*   role = (const int*)  d_in[3];
  const float* Wk = (const float*)d_in[4];  const float* bk  = (const float*)d_in[5];
  const float* Wa = (const float*)d_in[6];  const float* ba  = (const float*)d_in[7];
  const float* Wr = (const float*)d_in[8];  const float* br  = (const float*)d_in[9];
  const float* g_in = (const float*)d_in[10]; const float* b_in = (const float*)d_in[11];
  const float* Wb0 = (const float*)d_in[12]; const float* bb0 = (const float*)d_in[13];
  const float* g0  = (const float*)d_in[14]; const float* be0 = (const float*)d_in[15];
  const float* Wb1 = (const float*)d_in[16]; const float* bb1 = (const float*)d_in[17];
  const float* g1  = (const float*)d_in[18]; const float* be1 = (const float*)d_in[19];
  const float* Ww  = (const float*)d_in[20]; const float* bw  = (const float*)d_in[21];
  const float* Wq  = (const float*)d_in[22]; const float* bq  = (const float*)d_in[23];
  const float* Wad = (const float*)d_in[24]; const float* bad = (const float*)d_in[25];

  // workspace layout (bf16 elements)
  __bf16* ws = (__bf16*)d_ws;
  size_t oWk  = 0;
  size_t oWa  = oWk  + (size_t)128 * 256;
  size_t oWr  = oWa  + (size_t)256 * 256;
  size_t oWb0 = oWr  + (size_t)256 * 256;
  size_t oWb1 = oWb0 + (size_t)256 * 256;
  size_t oWw  = oWb1 + (size_t)256 * 256;
  size_t oWq  = oWw  + (size_t)256 * 1024;
  size_t oWad = oWq  + (size_t)256 * 1024;

  auto prep = [&](const float* src, size_t off, int K, int Nc) {
    int total = K * Nc;
    prep_weights<<<(total + 255) / 256, 256, 0, stream>>>(src, ws + off, K, Nc);
  };
  prep(Wk,  oWk,  128, 256);
  prep(Wa,  oWa,  256, 256);
  prep(Wr,  oWr,  256, 256);
  prep(Wb0, oWb0, 256, 256);
  prep(Wb1, oWb1, 256, 256);
  prep(Ww,  oWw,  256, 1024);
  prep(Wq,  oWq,  256, 1024);
  prep(Wad, oWad, 256, 64);

  float* out_logits = (float*)d_out;
  float* out_h      = out_logits + (size_t)N * 1024;
  float* out_addr   = out_h + (size_t)N * 256;

  apsg_main<<<N / 16, 256, 0, stream>>>(
      key, aux, res, role, bk, ba, br, g_in, b_in,
      bb0, g0, be0, bb1, g1, be1, bw, bq, bad,
      ws + oWk, ws + oWa, ws + oWr, ws + oWb0, ws + oWb1,
      ws + oWw, ws + oWq, ws + oWad,
      out_logits, out_h, out_addr);
}
